// MoELoRALinear_74208444940805
// MI455X (gfx1250) — matrix operand order repaired and checked by reading, not compiled, not run on hardware
//
#include <hip/hip_runtime.h>
#include <math.h>

typedef _Float16 f16;
typedef __attribute__((ext_vector_type(16))) _Float16 v16h;
typedef __attribute__((ext_vector_type(8)))  _Float16 v8h;
typedef __attribute__((ext_vector_type(8)))  float    v8f;

constexpr int kD   = 1024;
constexpr int kE   = 8;
constexpr int kR   = 8;
constexpr int kH   = 4;
constexpr int kB   = 8;
constexpr int kS   = 4096;
constexpr int kSkv = 512;
constexpr int kHD  = kD / kH;     // 256
constexpr int kER  = kE * kR;     // 64
constexpr int kM   = kB * kS;     // 32768
constexpr int kMkv = kB * kSkv;   // 4096
constexpr float kScaling = 16.0f / kR; // 2.0

// ---------------------------------------------------------------------------
// WMMA helpers (gfx1250, wave32). D = A(16x32) * B(32x16) + C, f16 in / f32 acc
// ---------------------------------------------------------------------------
__device__ __forceinline__ v8f wmma_f16(v16h a, v16h b, v8f c) {
  return __builtin_amdgcn_wmma_f32_16x16x32_f16(
      /*neg_a=*/false, a, /*neg_b=*/false, b,
      /*c_mod=*/(short)0, c, /*reuse_a=*/false, /*reuse_b=*/false);
}

// A-operand tile (16 rows x 32 K) from row-major [rows x ld] f16 matrix.
// ISA layout: lane<16 -> row=lane, K {0..7, 16..23}; lane>=16 -> row=lane-16, K {8..15, 24..31}
__device__ __forceinline__ v16h load_a_16x32(const f16* p, int ld, int row0, int k0) {
  int lane = threadIdx.x & 31;
  const f16* base = p + (size_t)(row0 + (lane & 15)) * ld + k0 + ((lane >> 4) << 3);
  v8h lo = *(const v8h*)(base);
  v8h hi = *(const v8h*)(base + 16);
  return __builtin_shufflevector(lo, hi, 0,1,2,3,4,5,6,7,8,9,10,11,12,13,14,15);
}

// B-operand tile (32 K x 16 cols) taken from a row-major (N x K) "weight" matrix W,
// so the WMMA computes A @ W^T.  ISA layout: lane<16 -> col=lane, K {0..7, 16..23};
// lane>=16 -> col=lane-16, K {8..15, 24..31}.
__device__ __forceinline__ v16h load_b_32x16(const f16* p, int ld, int n0, int k0) {
  int lane = threadIdx.x & 31;
  const f16* base = p + (size_t)(n0 + (lane & 15)) * ld + k0 + ((lane >> 4) << 3);
  v8h lo = *(const v8h*)(base);
  v8h hi = *(const v8h*)(base + 16);
  return __builtin_shufflevector(lo, hi, 0,1,2,3,4,5,6,7,8,9,10,11,12,13,14,15);
}

// ---------------------------------------------------------------------------
// Elementwise / reduction helpers
// ---------------------------------------------------------------------------
__global__ void k_cvt(const float* __restrict__ in, f16* __restrict__ out, int n) {
  int i = blockIdx.x * 256 + threadIdx.x;
  if (i < n) out[i] = (f16)in[i];
}

__global__ void k_mean(const float* __restrict__ x, float* __restrict__ xmean) {
  int i = blockIdx.x * 256 + threadIdx.x;   // i in [0, B*D)
  int b = i / kD, d = i % kD;
  const float* p = x + (size_t)b * kS * kD + d;
  float s = 0.f;
  for (int t = 0; t < kS; ++t) s += p[(size_t)t * kD];
  xmean[i] = s * (1.0f / kS);
}

__global__ void k_router(const float* __restrict__ xmean, const float* __restrict__ Wr,
                         const float* __restrict__ br, float* __restrict__ w) {
  __shared__ float lg[kB * kE];
  int t = threadIdx.x;                       // 64 threads
  if (t < kB * kE) {
    int b = t / kE, e = t % kE;
    const float* xm = xmean + b * kD;
    const float* wr = Wr + e * kD;
    float s = br[e];
    for (int d = 0; d < kD; ++d) s += xm[d] * wr[d];
    lg[t] = s;
  }
  __syncthreads();
  if (t < kB) {
    float mx = -1e30f;
    for (int e = 0; e < kE; ++e) mx = fmaxf(mx, lg[t * kE + e]);
    float sum = 0.f, ex[kE];
    for (int e = 0; e < kE; ++e) { ex[e] = expf(lg[t * kE + e] - mx); sum += ex[e]; }
    float inv = 1.0f / sum;
    for (int e = 0; e < kE; ++e) w[t * kE + e] = ex[e] * inv;
  }
}

// bw[b][o][e*R+r] = SCALING * w[b,e] * Bm[e,o,r]   (f16, (B x D x 64), ld=64)
__global__ void k_bw(const float* __restrict__ Bm, const float* __restrict__ w,
                     f16* __restrict__ bw) {
  int i = blockIdx.x * 256 + threadIdx.x;   // B*D*ER
  if (i >= kB * kD * kER) return;
  int b   = i / (kD * kER);
  int rem = i % (kD * kER);
  int o   = rem / kER;
  int er  = rem % kER;
  int e = er / kR, r = er % kR;
  bw[i] = (f16)(kScaling * w[b * kE + e] * Bm[(size_t)e * kD * kR + o * kR + r]);
}

// ---------------------------------------------------------------------------
// Generic WMMA GEMM:  C[m,n] = A[m,:] . W[n,:] (+ bias[n])
// Block = 128 threads (4 waves); block tile = 128 rows x 64 cols; wave tile 32x64.
// The 64x32 B-slab for each K-step is staged into LDS with CDNA5 async
// global->LDS copies (double buffered, drained with s_wait_asynccnt), so the
// next slab's copies overlap the current slab's WMMAs.
// MODE 0: f32 store; 1: f16 store; 2: f32 accumulate-add; 3: f16 transposed store
// blockIdx.z batches with strides sA/sW/sC (element strides).
// ---------------------------------------------------------------------------
template <int MODE>
__global__ void __launch_bounds__(128)
k_gemm(const f16* __restrict__ A, const f16* __restrict__ W,
       const float* __restrict__ bias, void* __restrict__ Cv,
       int K, int ldA, int ldW, int ldC,
       long long sA, long long sW, long long sC) {
  __shared__ alignas(16) f16 wbuf[2][64 * 32];   // 2 x 4 KB B-slab buffers
  int z = blockIdx.z;
  A += (size_t)z * sA;
  W += (size_t)z * sW;
  int wave = threadIdx.x >> 5;
  int row0 = blockIdx.y * 128 + wave * 32;
  int col0 = blockIdx.x * 64;
  int t = threadIdx.x;

  // Issue async copies of the 64-row x 32-K B slab (k..k+31) into wbuf[buf].
  // 256 chunks of 16B; 128 threads x 2 issues -> 2 ASYNCcnt per wave.
  auto stage = [&](int buf, int k) {
    #pragma unroll
    for (int i = 0; i < 2; ++i) {
      int c   = t + i * 128;
      int row = c >> 2;
      int ho  = (c & 3) * 8;                       // half offset within slab row
      const f16* g = W + (size_t)(col0 + row) * ldW + k + ho;
      unsigned lds = (unsigned)(uintptr_t)(&wbuf[buf][row * 32 + ho]);
      asm volatile("global_load_async_to_lds_b128 %0, %1, off"
                   :: "v"(lds), "v"(g) : "memory");
    }
  };

  v8f acc[2][4];
  #pragma unroll
  for (int i = 0; i < 2; ++i)
    #pragma unroll
    for (int j = 0; j < 4; ++j)
      acc[i][j] = (v8f){0.f,0.f,0.f,0.f,0.f,0.f,0.f,0.f};

  stage(0, 0);
  int buf = 0;
  for (int k = 0; k < K; k += 32, buf ^= 1) {
    if (k + 32 < K) {
      stage(buf ^ 1, k + 32);
      // 4 outstanding per wave; async loads retire in order, so <=2 left
      // means the current buffer's two copies have landed.
      asm volatile("s_wait_asynccnt 0x2" ::: "memory");
    } else {
      asm volatile("s_wait_asynccnt 0x0" ::: "memory");
    }
    __syncthreads();                      // slab visible to all 4 waves

    v16h a0 = load_a_16x32(A, ldA, row0,      k);
    v16h a1 = load_a_16x32(A, ldA, row0 + 16, k);
    if (k + 64 < K)                       // warm caches for the next A chunk
      __builtin_prefetch((const void*)(A + (size_t)row0 * ldA + k + 64), 0, 3);
    #pragma unroll
    for (int j = 0; j < 4; ++j) {
      v16h b = load_b_32x16(&wbuf[buf][0], 32, 16 * j, 0);
      acc[0][j] = wmma_f16(a0, b, acc[0][j]);
      acc[1][j] = wmma_f16(a1, b, acc[1][j]);
    }
    __syncthreads();                      // done reading before slab is reused
  }

  int lane = threadIdx.x & 31;
  int n  = lane & 15;
  int mh = lane >> 4;
  #pragma unroll
  for (int i = 0; i < 2; ++i) {
    #pragma unroll
    for (int j = 0; j < 4; ++j) {
      int cn = col0 + 16 * j + n;
      float bv = bias ? bias[cn] : 0.0f;
      #pragma unroll
      for (int r = 0; r < 8; ++r) {
        int m = row0 + i * 16 + mh * 8 + r;
        float val = acc[i][j][r] + bv;
        if constexpr (MODE == 0)
          ((float*)Cv)[(size_t)z * sC + (size_t)m * ldC + cn] = val;
        else if constexpr (MODE == 1)
          ((f16*)Cv)[(size_t)z * sC + (size_t)m * ldC + cn] = (f16)val;
        else if constexpr (MODE == 2)
          ((float*)Cv)[(size_t)z * sC + (size_t)m * ldC + cn] += val;
        else
          ((f16*)Cv)[(size_t)z * sC + (size_t)cn * ldC + m] = (f16)val;
      }
    }
  }
}

// ---------------------------------------------------------------------------
// Fused attention for one (b, head, 16-query tile).  Block = 128 thr (4 waves).
// scores = (Q Kt)/16 -> softmax -> O = P V.  V supplied transposed (D x Mkv).
// ---------------------------------------------------------------------------
__global__ void __launch_bounds__(128)
k_attn(const f16* __restrict__ qh, const f16* __restrict__ kh,
       const f16* __restrict__ vT, f16* __restrict__ oh) {
  __shared__ alignas(16) float sc[16 * kSkv];  // 32 KB
  __shared__ alignas(16) f16   pl[16 * kSkv];  // 16 KB
  int qt = blockIdx.x, hh = blockIdx.y, b = blockIdx.z;
  int wave = threadIdx.x >> 5;
  int lane = threadIdx.x & 31;
  int rowQ = b * kS + qt * 16;

  // Hoist the 16x256 Q tile (8 K-chunks) into registers.
  v16h aq[8];
  #pragma unroll
  for (int kk = 0; kk < 8; ++kk)
    aq[kk] = load_a_16x32(qh, kD, rowQ, hh * kHD + kk * 32);

  // Scores: each wave covers 8 of the 32 kv column tiles.
  for (int ct = wave; ct < kSkv / 16; ct += 4) {
    v8f acc = (v8f){0.f,0.f,0.f,0.f,0.f,0.f,0.f,0.f};
    #pragma unroll
    for (int kk = 0; kk < 8; ++kk) {
      v16h bb = load_b_32x16(kh, kD, b * kSkv + ct * 16, hh * kHD + kk * 32);
      acc = wmma_f16(aq[kk], bb, acc);
    }
    int n = lane & 15, mh = lane >> 4;
    #pragma unroll
    for (int r = 0; r < 8; ++r)
      sc[(mh * 8 + r) * kSkv + ct * 16 + n] = acc[r];
  }
  __syncthreads();

  // Row softmax (scale 1/sqrt(hd) = 1/16), write f16 P.
  if (threadIdx.x < 16) {
    int r = threadIdx.x;
    float mx = -1e30f;
    for (int c = 0; c < kSkv; ++c) mx = fmaxf(mx, sc[r * kSkv + c]);
    float sum = 0.f;
    for (int c = 0; c < kSkv; ++c) {
      float e = expf((sc[r * kSkv + c] - mx) * 0.0625f);
      sc[r * kSkv + c] = e;
      sum += e;
    }
    float inv = 1.0f / sum;
    for (int c = 0; c < kSkv; ++c)
      pl[r * kSkv + c] = (f16)(sc[r * kSkv + c] * inv);
  }
  __syncthreads();

  // O = P @ V : each wave owns 64 of the 256 head-dim columns.
  int nb = wave * 64;
  v8f acc[4];
  #pragma unroll
  for (int j = 0; j < 4; ++j)
    acc[j] = (v8f){0.f,0.f,0.f,0.f,0.f,0.f,0.f,0.f};
  for (int k = 0; k < kSkv; k += 32) {
    v16h a = load_a_16x32(pl, kSkv, 0, k);
    #pragma unroll
    for (int j = 0; j < 4; ++j) {
      v16h bb = load_b_32x16(vT, kMkv, hh * kHD + nb + 16 * j, b * kSkv + k);
      acc[j] = wmma_f16(a, bb, acc[j]);
    }
  }
  int n = lane & 15, mh = lane >> 4;
  #pragma unroll
  for (int j = 0; j < 4; ++j)
    #pragma unroll
    for (int r = 0; r < 8; ++r)
      oh[(size_t)(rowQ + mh * 8 + r) * kD + hh * kHD + nb + 16 * j + n] = (f16)acc[j][r];
}

// ---------------------------------------------------------------------------
// Host launcher
// ---------------------------------------------------------------------------
extern "C" void kernel_launch(void* const* d_in, const int* in_sizes, int n_in,
                              void* d_out, int out_size, void* d_ws, size_t ws_size,
                              hipStream_t stream) {
  const float* x       = (const float*)d_in[0];
  const float* geom    = (const float*)d_in[1];
  const float* W_lin   = (const float*)d_in[2];
  const float* b_lin   = (const float*)d_in[3];
  const float* W_rout  = (const float*)d_in[4];
  const float* b_rout  = (const float*)d_in[5];
  const float* A_in    = (const float*)d_in[6];   // (E,R,D) == A_cat (64 x 1024)
  const float* Bm      = (const float*)d_in[7];   // (E,D,R)
  const float* ipw     = (const float*)d_in[8];   // (3D, D)
  const float* ipb     = (const float*)d_in[9];
  const float* opw     = (const float*)d_in[10];
  const float* opb     = (const float*)d_in[11];
  float* out = (float*)d_out;
  (void)in_sizes; (void)n_in; (void)out_size; (void)ws_size;

  char* p = (char*)d_ws;
  auto take = [&](size_t bytes) { void* r = (void*)p; p += (bytes + 255) & ~(size_t)255; return r; };
  f16*   x_h    = (f16*)take((size_t)kM * kD * 2);     // reused as attention output later
  f16*   geom_h = (f16*)take((size_t)kMkv * kD * 2);
  f16*   wlin_h = (f16*)take((size_t)kD * kD * 2);
  f16*   winp_h = (f16*)take((size_t)3 * kD * kD * 2);
  f16*   wo_h   = (f16*)take((size_t)kD * kD * 2);
  f16*   acat_h = (f16*)take((size_t)kER * kD * 2);
  f16*   q_h    = (f16*)take((size_t)kM * kD * 2);
  f16*   k_h    = (f16*)take((size_t)kMkv * kD * 2);
  f16*   vT_h   = (f16*)take((size_t)kD * kMkv * 2);   // transposed V
  f16*   h_h    = (f16*)take((size_t)kM * kER * 2);
  f16*   bw_h   = (f16*)take((size_t)kB * kD * kER * 2);
  float* xmean  = (float*)take((size_t)kB * kD * 4);
  float* wrt    = (float*)take((size_t)kB * kE * 4);
  f16*   o_h    = x_h;  // x dead after Q/base/h GEMMs -> reuse for attention output

  auto cvt = [&](const float* src, f16* dst, size_t n) {
    k_cvt<<<dim3((unsigned)((n + 255) / 256)), dim3(256), 0, stream>>>(src, dst, (int)n);
  };
  cvt(x, x_h, (size_t)kM * kD);
  cvt(geom, geom_h, (size_t)kMkv * kD);
  cvt(W_lin, wlin_h, (size_t)kD * kD);
  cvt(ipw, winp_h, (size_t)3 * kD * kD);
  cvt(opw, wo_h, (size_t)kD * kD);
  cvt(A_in, acat_h, (size_t)kER * kD);

  k_mean<<<dim3(kB * kD / 256), dim3(256), 0, stream>>>(x, xmean);
  k_router<<<dim3(1), dim3(64), 0, stream>>>(xmean, W_rout, b_rout, wrt);
  k_bw<<<dim3(kB * kD * kER / 256), dim3(256), 0, stream>>>(Bm, wrt, bw_h);

  dim3 blk(128);
  // base_out = x @ W_lin^T + b_lin  -> d_out (f32)
  k_gemm<0><<<dim3(kD / 64, kM / 128, 1), blk, 0, stream>>>(
      x_h, wlin_h, b_lin, out, kD, kD, kD, kD, 0, 0, 0);
  // q = x @ Wq^T + bq -> f16
  k_gemm<1><<<dim3(kD / 64, kM / 128, 1), blk, 0, stream>>>(
      x_h, winp_h, ipb, q_h, kD, kD, kD, kD, 0, 0, 0);
  // k = geom @ Wk^T + bk -> f16
  k_gemm<1><<<dim3(kD / 64, kMkv / 128, 1), blk, 0, stream>>>(
      geom_h, winp_h + (size_t)kD * kD, ipb + kD, k_h, kD, kD, kD, kD, 0, 0, 0);
  // vT = (geom @ Wv^T + bv)^T -> f16 transposed (D x Mkv)
  k_gemm<3><<<dim3(kD / 64, kMkv / 128, 1), blk, 0, stream>>>(
      geom_h, winp_h + (size_t)2 * kD * kD, ipb + 2 * kD, vT_h, kD, kD, kD, kMkv, 0, 0, 0);
  // h = x @ A_cat^T -> (M x 64) f16
  k_gemm<1><<<dim3(1, kM / 128, 1), blk, 0, stream>>>(
      x_h, acat_h, nullptr, h_h, kD, kD, kD, kER, 0, 0, 0);
  // expert_out: per batch b, d_out += h[b] @ Bw[b]^T  (K = 64)
  k_gemm<2><<<dim3(kD / 64, kS / 128, kB), blk, 0, stream>>>(
      h_h, bw_h, nullptr, out, kER, kER, kER, kD,
      (long long)kS * kER, (long long)kD * kER, (long long)kS * kD);
  // attention -> o_h (f16, reuses x_h)
  k_attn<<<dim3(kS / 16, kH, kB), blk, 0, stream>>>(q_h, k_h, vT_h, o_h);
  // d_out += o @ Wo^T + bo
  k_gemm<2><<<dim3(kD / 64, kM / 128, 1), blk, 0, stream>>>(
      o_h, wo_h, opb, out, kD, kD, kD, kD, 0, 0, 0);
}
